// RNNModel_88639535055428
// MI455X (gfx1250) — compile-verified, matrix operand order checked by reading, not run on hardware
//
#include <hip/hip_runtime.h>

typedef __attribute__((ext_vector_type(16))) _Float16 v16h;
typedef __attribute__((ext_vector_type(8)))  float    v8f;

#define T_STEPS 128
#define BATCH   32
#define NTOKENS 32000
#define NINP    1024
#define NHID    2048
#define NBLK    8
#define TOPK    4
#define BS      256      // NHID / NBLK
#define GATES   1024     // 4 * BS
#define DK      64
#define CH      4
#define CDK     32
#define CDV     32
#define INV_SQRT_DK  0.125f
#define INV_SQRT_CDK 0.17677669529663687f

__device__ __forceinline__ float sigf(float x) { return 1.0f / (1.0f + __expf(-x)); }

// --- CDNA5 async global->LDS copy (ASYNCcnt-tracked, ISA §10.7 / §15.18.3) --
__device__ __forceinline__ unsigned ldsAddr(const void* p) {
    // flat LDS address: low 32 bits are the LDS byte offset (ISA §10.2 aperture)
    return (unsigned)(unsigned long long)p;
}
__device__ __forceinline__ void asyncLoadB128(unsigned lds, const void* g) {
    asm volatile("global_load_async_to_lds_b128 %0, %1, off"
                 :: "v"(lds), "v"(g) : "memory");
}
__device__ __forceinline__ void waitAsync0() {
    asm volatile("s_wait_asynccnt 0" ::: "memory");
}

// ---------------------------------------------------------------------------
// Async double-buffered GEMM: C = A · Bᵀ, f16 operands, fp32 WMMA accumulate.
//   A: f16 [M][K] (lda), B: f16 [N][K] (ldb)
//   OM==0: C fp32 [M][N] (+bias)   OM==1: C f16 [M][N]   OM==2: C f16 [N][M]
// Contract: M % 64 == 0, N % 128 == 0, K % 32 == 0 (no guards anywhere).
// Block tile 64x128, BK=32, 256 thr = 8 waves (4 M-subtiles x 2 N-halves),
// each wave: 1 A fragment x 4 B fragments -> 4 WMMAs per K-chunk.
// Staging: 3x global_load_async_to_lds_b128 per thread per chunk into the
// spare LDS buffer while WMMAs consume the current one; s_wait_asynccnt 0 +
// barrier closes each stage. LDS tiles are pre-permuted into fragment order.
// ---------------------------------------------------------------------------
template <int OM>
__global__ __launch_bounds__(256) void gemm_h_async(
    int M, int N, int K,
    const _Float16* __restrict__ A, int lda, long long strideA,
    const _Float16* __restrict__ B, int ldb, long long strideB,
    void* __restrict__ Cv, int ldc, long long strideC,
    const float* __restrict__ bias)
{
    const int bz = blockIdx.z;
    A += (long long)bz * strideA;
    B += (long long)bz * strideB;

    const int row0 = blockIdx.y * 64;
    const int col0 = blockIdx.x * 128;

    __shared__ __align__(32) _Float16 AL[2][4 * 32 * 16];   // 2 x 4KB
    __shared__ __align__(32) _Float16 BL[2][8 * 32 * 16];   // 2 x 8KB

    const int tid  = threadIdx.x;
    const int lane = tid & 31;
    const int wave = tid >> 5;
    const int wm   = wave >> 1;   // 0..3
    const int wn   = wave & 1;    // 0..1

    // staging task: (sub-tile st, fragment lane fl, k-group g)
    const int st = tid >> 6;          // 0..3
    const int fl = (tid >> 1) & 31;   // 0..31
    const int g  = tid & 1;           // 0..1
    const int koff = g * 16 + ((fl >> 4) << 3);
    const long long aRow  = (long long)(row0 + st * 16 + (fl & 15)) * lda;
    const long long bRow0 = (long long)(col0 + st * 16 + (fl & 15)) * ldb;
    const long long bRow1 = (long long)(col0 + (st + 4) * 16 + (fl & 15)) * ldb;
    const int aLds  = (st * 32 + fl) * 16 + g * 8;
    const int bLds0 = (st * 32 + fl) * 16 + g * 8;
    const int bLds1 = ((st + 4) * 32 + fl) * 16 + g * 8;

    v8f acc0 = {}, acc1 = {}, acc2 = {}, acc3 = {};

    auto issue = [&](int k0, int buf) {
        asyncLoadB128(ldsAddr(&AL[buf][aLds]),  A + aRow  + k0 + koff);
        asyncLoadB128(ldsAddr(&BL[buf][bLds0]), B + bRow0 + k0 + koff);
        asyncLoadB128(ldsAddr(&BL[buf][bLds1]), B + bRow1 + k0 + koff);
    };

    issue(0, 0);
    waitAsync0();
    __syncthreads();

    int buf = 0;
    for (int k0 = 0; k0 < K; k0 += 32) {
        if (k0 + 32 < K) issue(k0 + 32, buf ^ 1);   // prefetch next chunk

        const v16h a  = *(const v16h*)&AL[buf][(wm * 32 + lane) * 16];
        const v16h b0 = *(const v16h*)&BL[buf][((wn * 4 + 0) * 32 + lane) * 16];
        const v16h b1 = *(const v16h*)&BL[buf][((wn * 4 + 1) * 32 + lane) * 16];
        const v16h b2 = *(const v16h*)&BL[buf][((wn * 4 + 2) * 32 + lane) * 16];
        const v16h b3 = *(const v16h*)&BL[buf][((wn * 4 + 3) * 32 + lane) * 16];
        acc0 = __builtin_amdgcn_wmma_f32_16x16x32_f16(false, a, false, b0, (short)0, acc0, false, false);
        acc1 = __builtin_amdgcn_wmma_f32_16x16x32_f16(false, a, false, b1, (short)0, acc1, false, false);
        acc2 = __builtin_amdgcn_wmma_f32_16x16x32_f16(false, a, false, b2, (short)0, acc2, false, false);
        acc3 = __builtin_amdgcn_wmma_f32_16x16x32_f16(false, a, false, b3, (short)0, acc3, false, false);

        waitAsync0();
        __syncthreads();
        buf ^= 1;
    }

    const int mbase = row0 + wm * 16 + ((lane >> 4) << 3);
    v8f accs[4] = {acc0, acc1, acc2, acc3};
    if (OM == 0) {
        float* C = (float*)Cv + (long long)bz * strideC;
        #pragma unroll
        for (int i = 0; i < 4; ++i) {
            const int col = col0 + wn * 64 + i * 16 + (lane & 15);
            const float bb = bias ? bias[col] : 0.0f;
            #pragma unroll
            for (int r = 0; r < 8; ++r)
                C[(long long)(mbase + r) * ldc + col] = accs[i][r] + bb;
        }
    } else if (OM == 1) {
        _Float16* C = (_Float16*)Cv + (long long)bz * strideC;
        #pragma unroll
        for (int i = 0; i < 4; ++i) {
            const int col = col0 + wn * 64 + i * 16 + (lane & 15);
            #pragma unroll
            for (int r = 0; r < 8; ++r)
                C[(long long)(mbase + r) * ldc + col] = (_Float16)accs[i][r];
        }
    } else { // OM == 2: f16 transposed [N][M]; 8 rows contiguous -> b128 store
        _Float16* C = (_Float16*)Cv + (long long)bz * strideC;
        #pragma unroll
        for (int i = 0; i < 4; ++i) {
            const int col = col0 + wn * 64 + i * 16 + (lane & 15);
            __align__(16) _Float16 t[8];
            #pragma unroll
            for (int r = 0; r < 8; ++r) t[r] = (_Float16)accs[i][r];
            *(uint4*)&C[(long long)col * ldc + mbase] = *(const uint4*)t;
        }
    }
}

// ---------------------------------------------------------------------------
// Guarded synchronous GEMM (64x64 tile) for small/ragged shapes:
// used for Kt (N=64) and the per-step h@Wh (M=32). fp32 output only.
// ---------------------------------------------------------------------------
__global__ __launch_bounds__(256) void gemm_h_guard(
    int M, int N, int K,
    const _Float16* __restrict__ A, int lda, long long strideA,
    const _Float16* __restrict__ B, int ldb, long long strideB,
    float* __restrict__ C, int ldc, long long strideC)
{
    const int bz = blockIdx.z;
    A += (long long)bz * strideA;
    B += (long long)bz * strideB;
    C += (long long)bz * strideC;

    const int row0 = blockIdx.y * 64;
    const int col0 = blockIdx.x * 64;

    __shared__ __align__(32) _Float16 AL[4 * 32 * 16];
    __shared__ __align__(32) _Float16 BL[4 * 32 * 16];

    const int tid  = threadIdx.x;
    const int lane = tid & 31;
    const int wave = tid >> 5;
    const int wm   = wave >> 1;   // 0..3
    const int wn   = wave & 1;    // 0..1

    const int st = tid >> 6;
    const int fl = (tid >> 1) & 31;
    const int g  = tid & 1;
    const int arow = row0 + st * 16 + (fl & 15);
    const int brow = col0 + st * 16 + (fl & 15);
    const int arowc = arow < M ? arow : (M - 1);
    const int browc = brow < N ? brow : (N - 1);
    const int koff = g * 16 + ((fl >> 4) << 3);
    const int ldsOff = (st * 32 + fl) * 16 + g * 8;

    v8f c0 = {}; v8f c1 = {};

    for (int k0 = 0; k0 < K; k0 += 32) {
        {
            uint4 v = *(const uint4*)(A + (long long)arowc * lda + (k0 + koff));
            if (arow >= M) { v.x = 0u; v.y = 0u; v.z = 0u; v.w = 0u; }
            *(uint4*)&AL[ldsOff] = v;
        }
        {
            uint4 v = *(const uint4*)(B + (long long)browc * ldb + (k0 + koff));
            if (brow >= N) { v.x = 0u; v.y = 0u; v.z = 0u; v.w = 0u; }
            *(uint4*)&BL[ldsOff] = v;
        }
        __syncthreads();

        const v16h a  = *(const v16h*)&AL[(wm * 32 + lane) * 16];
        const v16h b0 = *(const v16h*)&BL[((wn * 2 + 0) * 32 + lane) * 16];
        const v16h b1 = *(const v16h*)&BL[((wn * 2 + 1) * 32 + lane) * 16];
        c0 = __builtin_amdgcn_wmma_f32_16x16x32_f16(false, a, false, b0, (short)0, c0, false, false);
        c1 = __builtin_amdgcn_wmma_f32_16x16x32_f16(false, a, false, b1, (short)0, c1, false, false);
        __syncthreads();
    }

    const int mbase = row0 + wm * 16 + ((lane >> 4) << 3);
    const int nA = col0 + wn * 32 + (lane & 15);
    const int nB = nA + 16;
    #pragma unroll
    for (int r = 0; r < 8; ++r) {
        const int row = mbase + r;
        if (row < M) {
            if (nA < N) C[(long long)row * ldc + nA] = c0[r];
            if (nB < N) C[(long long)row * ldc + nB] = c1[r];
        }
    }
}

// ---------------------------------------------------------------------------
// fp32 -> f16 elementwise cast (vectorized)
// ---------------------------------------------------------------------------
__global__ void cast_f16(const float* __restrict__ in, _Float16* __restrict__ out,
                         long long count)
{
    const long long i = ((long long)blockIdx.x * blockDim.x + threadIdx.x) * 4;
    if (i + 3 < count) {
        const float4 v = *(const float4*)(in + i);
        __align__(8) _Float16 h[4] = {(_Float16)v.x, (_Float16)v.y, (_Float16)v.z, (_Float16)v.w};
        *(uint2*)(out + i) = *(const uint2*)h;
    } else {
        for (long long j = i; j < count; ++j) out[j] = (_Float16)in[j];
    }
}

// ---------------------------------------------------------------------------
// fp32 [K][N] -> f16 [N][K] transposed cast (batched, LDS tile transpose)
// ---------------------------------------------------------------------------
__global__ __launch_bounds__(256) void cast_transpose_f16(
    const float* __restrict__ in, _Float16* __restrict__ out, int K, int N)
{
    __shared__ float tile[32][33];
    const long long base = (long long)blockIdx.z * K * N;
    const int k0 = blockIdx.y * 32, n0 = blockIdx.x * 32;
    const int c = threadIdx.x & 31, r0 = threadIdx.x >> 5;
    for (int rr = r0; rr < 32; rr += 8) {
        const int k = k0 + rr, n = n0 + c;
        tile[rr][c] = (k < K && n < N) ? in[base + (long long)k * N + n] : 0.0f;
    }
    __syncthreads();
    for (int rr = r0; rr < 32; rr += 8) {
        const int n = n0 + rr, k = k0 + c;
        if (n < N && k < K) out[base + (long long)n * K + k] = (_Float16)tile[c][rr];
    }
}

// ---------------------------------------------------------------------------
// Embedding gather, directly to f16: Xh[r,:] = (f16)emb[tokens[r],:]
// ---------------------------------------------------------------------------
__global__ void gather_embed_f16(const int* __restrict__ tokens,
                                 const float* __restrict__ emb,
                                 _Float16* __restrict__ X)
{
    const int r = blockIdx.x;
    const float* src = emb + (long long)tokens[r] * NINP;
    _Float16* dst = X + (long long)r * NINP;
    for (int i = threadIdx.x * 4; i < NINP; i += blockDim.x * 4) {
        const float4 v = *(const float4*)(src + i);
        __align__(8) _Float16 h[4] = {(_Float16)v.x, (_Float16)v.y, (_Float16)v.z, (_Float16)v.w};
        *(uint2*)(dst + i) = *(const uint2*)h;
    }
}

// ---------------------------------------------------------------------------
// state init: h = hx0, c = cx0, h16 = (f16)hx0
// ---------------------------------------------------------------------------
__global__ void init_state(const float* __restrict__ hx0, const float* __restrict__ cx0,
                           float* __restrict__ h, float* __restrict__ c,
                           _Float16* __restrict__ h16)
{
    const int i = blockIdx.x * 256 + threadIdx.x;
    if (i < BATCH * NHID) {
        const float hv = hx0[i];
        h[i] = hv; c[i] = cx0[i]; h16[i] = (_Float16)hv;
    }
}

// ---------------------------------------------------------------------------
// Per-step input attention + jax.lax.top_k-compatible mask
// ---------------------------------------------------------------------------
__global__ __launch_bounds__(256) void step_attn(
    const float* __restrict__ h, const float* __restrict__ Wq,
    const float* __restrict__ Kt, int t,
    float* __restrict__ att0, float* __restrict__ mask)
{
    __shared__ float kv[DK];
    __shared__ float wkq[BS];
    __shared__ float sc[NBLK];
    const int b = blockIdx.x;
    const int tid = threadIdx.x;
    const long long r = (long long)t * BATCH + b;
    if (tid < DK) kv[tid] = Kt[r * DK + tid];
    __syncthreads();
    {   // wkq[d] = sum_k Wq[d,k] * k_t[b,k]
        float s = 0.f;
        const float* wr = Wq + tid * DK;
        #pragma unroll
        for (int k = 0; k < DK; ++k) s += wr[k] * kv[k];
        wkq[tid] = s;
    }
    __syncthreads();
    const int n = tid >> 5, lane = tid & 31;
    float p = 0.f;
    const float* hb = h + (long long)b * NHID + n * BS;
    for (int d = lane; d < BS; d += 32) p += hb[d] * wkq[d];
    for (int off = 16; off > 0; off >>= 1) p += __shfl_down(p, off, 32);
    if (lane == 0) sc[n] = p * INV_SQRT_DK;
    __syncthreads();
    if (tid < NBLK) {
        const float sn = sc[tid];
        int rank = 0;   // rank by (att1 desc, index asc) == (score asc, index asc)
        for (int m = 0; m < NBLK; ++m) {
            if (m == tid) continue;
            const float sm = sc[m];
            if (sm < sn || (sm == sn && m < tid)) rank++;
        }
        mask[b * NBLK + tid] = (rank < (NBLK - TOPK)) ? 0.0f : 1.0f;
        att0[b * NBLK + tid] = sigf(sn);
    }
}

// ---------------------------------------------------------------------------
// LSTM pointwise: gates = att0*Z[t] + (h@Wh) + b_lstm -> c_new, h_pre
// ---------------------------------------------------------------------------
__global__ __launch_bounds__(256) void step_lstm(
    const float* __restrict__ gWh, const _Float16* __restrict__ Z, int t,
    const float* __restrict__ att0, const float* __restrict__ b_lstm,
    const float* __restrict__ c, float* __restrict__ hn, float* __restrict__ cn)
{
    const int bn = blockIdx.x;          // (b, n)
    const int b = bn >> 3, n = bn & 7;
    const int j = threadIdx.x;          // 0..255
    const long long r = (long long)t * BATCH + b;
    const float a0 = att0[b * NBLK + n];
    const float* gw = gWh + (long long)b * (NBLK * GATES) + n * GATES;
    const _Float16* zz = Z + ((long long)n * (T_STEPS * BATCH) + r) * GATES;
    const float* bl = b_lstm + n * GATES;
    const float gi = a0 * (float)zz[j]       + gw[j]       + bl[j];
    const float gf = a0 * (float)zz[256 + j] + gw[256 + j] + bl[256 + j];
    const float gg = a0 * (float)zz[512 + j] + gw[512 + j] + bl[512 + j];
    const float go = a0 * (float)zz[768 + j] + gw[768 + j] + bl[768 + j];
    const long long o = (long long)b * NHID + n * BS + j;
    const float cnew = sigf(gf) * c[o] + sigf(gi) * tanhf(gg);
    cn[o] = cnew;
    hn[o] = sigf(go) * tanhf(cnew);
}

// ---------------------------------------------------------------------------
// Communication attention + masked state update (also maintains f16 shadows)
// ---------------------------------------------------------------------------
__global__ __launch_bounds__(256) void step_comm(
    const float* __restrict__ hn, const float* __restrict__ cn,
    const float* __restrict__ Wcq, const float* __restrict__ Wck,
    const float* __restrict__ Wcv, const float* __restrict__ Wco,
    const float* __restrict__ mask,
    float* __restrict__ h, float* __restrict__ c,
    _Float16* __restrict__ h16, _Float16* __restrict__ outs16, int t)
{
    __shared__ float hs[NHID];
    __shared__ float cq[NBLK][CH * CDK];
    __shared__ float ck[NBLK][CH * CDK];
    __shared__ float cv[NBLK][CH * CDV];
    __shared__ float catt[CH][NBLK][NBLK];
    __shared__ float cm[NBLK][CH * CDV];
    const int b = blockIdx.x, tid = threadIdx.x;
    for (int i = tid; i < NHID; i += 256) hs[i] = hn[(long long)b * NHID + i];
    __syncthreads();
    for (int o = tid; o < NBLK * (CH * CDK); o += 256) {
        const int n = o >> 7, u = o & 127;
        const float* hr = &hs[n * BS];
        float sq = 0, sk = 0, sv = 0;
        for (int d = 0; d < BS; ++d) {
            const float hv = hr[d];
            sq += hv * Wcq[d * (CH * CDK) + u];
            sk += hv * Wck[d * (CH * CDK) + u];
            sv += hv * Wcv[d * (CH * CDV) + u];
        }
        cq[n][u] = sq; ck[n][u] = sk; cv[n][u] = sv;
    }
    __syncthreads();
    if (tid < CH * NBLK * NBLK) {
        const int hh = tid >> 6, n = (tid >> 3) & 7, m = tid & 7;
        float s = 0;
        #pragma unroll
        for (int v = 0; v < CDK; ++v) s += cq[n][hh * CDK + v] * ck[m][hh * CDK + v];
        catt[hh][n][m] = s * INV_SQRT_CDK;
    }
    __syncthreads();
    if (tid < CH * NBLK) {
        const int hh = tid >> 3, n = tid & 7;
        float mx = -1e30f;
        for (int m = 0; m < NBLK; ++m) mx = fmaxf(mx, catt[hh][n][m]);
        float e[NBLK], sum = 0;
        for (int m = 0; m < NBLK; ++m) { e[m] = __expf(catt[hh][n][m] - mx); sum += e[m]; }
        const float inv = 1.0f / sum;
        for (int m = 0; m < NBLK; ++m) catt[hh][n][m] = e[m] * inv;
    }
    __syncthreads();
    for (int o = tid; o < NBLK * (CH * CDV); o += 256) {
        const int n = o >> 7, u = o & 127, hh = u >> 5;
        float s = 0;
        #pragma unroll
        for (int m = 0; m < NBLK; ++m) s += catt[hh][n][m] * cv[m][u];
        cm[n][u] = s;
    }
    __syncthreads();
    for (int o = tid; o < NHID; o += 256) {
        const int n = o >> 8, d = o & 255;
        float s = 0;
        for (int u = 0; u < CH * CDV; ++u) s += cm[n][u] * Wco[u * BS + d];
        const float hnew = hs[o] + s;
        const float mk = mask[b * NBLK + n];
        const long long idx = (long long)b * NHID + o;
        const float hout = mk * hnew + (1.f - mk) * h[idx];
        const float cout = mk * cn[idx] + (1.f - mk) * c[idx];
        h[idx] = hout;
        c[idx] = cout;
        h16[idx] = (_Float16)hout;
        outs16[((long long)t * BATCH + b) * NHID + o] = (_Float16)hout;
    }
}

// ---------------------------------------------------------------------------
extern "C" void kernel_launch(void* const* d_in, const int* in_sizes, int n_in,
                              void* d_out, int out_size, void* d_ws, size_t ws_size,
                              hipStream_t stream)
{
    const int*   tokens = (const int*)  d_in[0];
    const float* hx0    = (const float*)d_in[1];
    const float* cx0    = (const float*)d_in[2];
    const float* emb    = (const float*)d_in[3];
    const float* Wq     = (const float*)d_in[4];
    const float* Wk     = (const float*)d_in[5];
    const float* Wv     = (const float*)d_in[6];
    const float* Wi     = (const float*)d_in[7];
    const float* Wh     = (const float*)d_in[8];
    const float* b_lstm = (const float*)d_in[9];
    const float* Wcq    = (const float*)d_in[10];
    const float* Wck    = (const float*)d_in[11];
    const float* Wcv    = (const float*)d_in[12];
    const float* Wco    = (const float*)d_in[13];
    const float* Wdec   = (const float*)d_in[14];
    const float* bdec   = (const float*)d_in[15];

    char* ws = (char*)d_ws;
    size_t off = 0;
    auto alloc = [&](size_t bytes) -> char* {
        char* p = ws + off; off += (bytes + 255) & ~(size_t)255; return p;
    };
    typedef _Float16 F16;
    F16*   Xh     = (F16*)alloc(sizeof(F16) * (size_t)T_STEPS * BATCH * NINP);
    F16*   WkT    = (F16*)alloc(sizeof(F16) * (size_t)DK * NINP);
    F16*   Wvh    = (F16*)alloc(sizeof(F16) * (size_t)NINP * GATES);
    F16*   WiT    = (F16*)alloc(sizeof(F16) * (size_t)NBLK * GATES * NINP);
    F16*   WhT    = (F16*)alloc(sizeof(F16) * (size_t)NBLK * GATES * BS);
    F16*   WfoldT = (F16*)alloc(sizeof(F16) * (size_t)NBLK * GATES * NINP);
    F16*   Wdech  = (F16*)alloc(sizeof(F16) * (size_t)NTOKENS * NHID);
    F16*   Zh     = (F16*)alloc(sizeof(F16) * (size_t)NBLK * T_STEPS * BATCH * GATES);
    F16*   Outs16 = (F16*)alloc(sizeof(F16) * (size_t)T_STEPS * BATCH * NHID);
    F16*   h16    = (F16*)alloc(sizeof(F16) * BATCH * NHID);
    float* Kt     = (float*)alloc(sizeof(float) * (size_t)T_STEPS * BATCH * DK);
    float* hst    = (float*)alloc(sizeof(float) * BATCH * NHID);
    float* cst    = (float*)alloc(sizeof(float) * BATCH * NHID);
    float* hn     = (float*)alloc(sizeof(float) * BATCH * NHID);
    float* cn     = (float*)alloc(sizeof(float) * BATCH * NHID);
    float* att0v  = (float*)alloc(sizeof(float) * BATCH * NBLK);
    float* maskv  = (float*)alloc(sizeof(float) * BATCH * NBLK);
    float* gWh    = (float*)alloc(sizeof(float) * BATCH * NBLK * GATES);

    // --- per-call casts / transposes to f16 --------------------------------
    init_state<<<(BATCH * NHID + 255) / 256, 256, 0, stream>>>(hx0, cx0, hst, cst, h16);
    gather_embed_f16<<<T_STEPS * BATCH, 256, 0, stream>>>(tokens, emb, Xh);
    cast_f16<<<(int)(((size_t)NINP * GATES / 4 + 255) / 256), 256, 0, stream>>>(
        Wv, Wvh, (long long)NINP * GATES);
    cast_f16<<<(int)(((size_t)NTOKENS * NHID / 4 + 255) / 256), 256, 0, stream>>>(
        Wdec, Wdech, (long long)NTOKENS * NHID);
    cast_transpose_f16<<<dim3(DK / 32, NINP / 32, 1), 256, 0, stream>>>(Wk, WkT, NINP, DK);
    cast_transpose_f16<<<dim3(GATES / 32, NINP / 32, NBLK), 256, 0, stream>>>(Wi, WiT, NINP, GATES);
    cast_transpose_f16<<<dim3(GATES / 32, BS / 32, NBLK), 256, 0, stream>>>(Wh, WhT, BS, GATES);

    // --- token-dependent precompute (hoisted out of the scan) --------------
    // Kt = Xh · WkTᵀ  (4096 x 64, K=1024) -> fp32   [guarded path, N=64]
    gemm_h_guard<<<dim3(1, 64, 1), 256, 0, stream>>>(
        T_STEPS * BATCH, DK, NINP, Xh, NINP, 0, WkT, NINP, 0, Kt, DK, 0);
    // WfoldT[n] = (Wv · Wi[n])ᵀ  -> f16 [GATES][NINP]
    gemm_h_async<2><<<dim3(GATES / 128, NINP / 64, NBLK), 256, 0, stream>>>(
        NINP, GATES, NINP, Wvh, NINP, 0, WiT, NINP, (long long)GATES * NINP,
        WfoldT, NINP, (long long)GATES * NINP, nullptr);
    // Z[n] = Xh · WfoldT[n]ᵀ  -> f16 [T*B][GATES]
    gemm_h_async<1><<<dim3(GATES / 128, (T_STEPS * BATCH) / 64, NBLK), 256, 0, stream>>>(
        T_STEPS * BATCH, GATES, NINP, Xh, NINP, 0,
        WfoldT, NINP, (long long)GATES * NINP,
        Zh, GATES, (long long)T_STEPS * BATCH * GATES, nullptr);

    // --- sequential recurrence ---------------------------------------------
    for (int t = 0; t < T_STEPS; ++t) {
        step_attn<<<BATCH, 256, 0, stream>>>(hst, Wq, Kt, t, att0v, maskv);
        // gWh[b, n*1024+j] = hb[b,n,:] @ Wh[n]  (batched over NBLK, M=32)
        gemm_h_guard<<<dim3(GATES / 64, 1, NBLK), 256, 0, stream>>>(
            BATCH, GATES, BS, h16, NHID, (long long)BS,
            WhT, BS, (long long)GATES * BS,
            gWh, NBLK * GATES, (long long)GATES);
        step_lstm<<<BATCH * NBLK, 256, 0, stream>>>(gWh, Zh, t, att0v, b_lstm, cst, hn, cn);
        step_comm<<<BATCH, 256, 0, stream>>>(hn, cn, Wcq, Wck, Wcv, Wco, maskv,
                                             hst, cst, h16, Outs16, t);
    }

    // --- decoder: d_out = Outs · Wdecᵀ + bdec  (4096 x 32000, K=2048) ------
    gemm_h_async<0><<<dim3(NTOKENS / 128, (T_STEPS * BATCH) / 64, 1), 256, 0, stream>>>(
        T_STEPS * BATCH, NTOKENS, NHID, Outs16, NHID, 0,
        Wdech, NHID, 0, (float*)d_out, NTOKENS, 0, bdec);

    (void)in_sizes; (void)n_in; (void)out_size; (void)ws_size;
}
